// MambaFFBlock_63797444214814
// MI455X (gfx1250) — matrix operand order repaired and checked by reading, not compiled, not run on hardware
//
#include <hip/hip_runtime.h>
#include <hip/hip_bf16.h>

// ---------------------------------------------------------------------------
// Problem constants (from reference)
// ---------------------------------------------------------------------------
constexpr int BATCH  = 2;
constexpr int LSEQ   = 2048;
constexpr int DMODEL = 1024;
constexpr int DINNER = 2048;
constexpr int TWOE   = 2 * DINNER;   // 4096
constexpr int DSTATE = 16;
constexpr int DCONV  = 4;
constexpr int DTRANK = 64;
constexpr int NXDBL  = DTRANK + 2 * DSTATE;  // 96
constexpr int FFH    = 4 * DMODEL;   // 4096
constexpr int ROWS   = BATCH * LSEQ; // 4096
constexpr int NCHUNK = 64;           // scan chunks
constexpr int CLEN   = 32;           // timesteps per chunk (NCHUNK*CLEN == LSEQ)

typedef __attribute__((ext_vector_type(16))) __bf16 v16bf;
typedef __attribute__((ext_vector_type(8)))  __bf16 v8bf;
typedef __attribute__((ext_vector_type(8)))  float  v8f;

// ---------------------------------------------------------------------------
// f32 -> bf16 conversion (weights)
// ---------------------------------------------------------------------------
__global__ __launch_bounds__(256)
void cvt_bf16_kernel(const float* __restrict__ in, __bf16* __restrict__ out, int n) {
    int i = blockIdx.x * 256 + threadIdx.x;
    if (i < n) out[i] = (__bf16)in[i];
}

// ---------------------------------------------------------------------------
// LayerNorm over D=1024, one block (8 wave32) per row, bf16 output
// ---------------------------------------------------------------------------
__global__ __launch_bounds__(256)
void ln_bf16_kernel(const float* __restrict__ x, const float* __restrict__ g,
                    const float* __restrict__ be, __bf16* __restrict__ out) {
    const int D = DMODEL;
    __shared__ float sred[8];
    const float* xr = x + (size_t)blockIdx.x * D;
    float v[4];
#pragma unroll
    for (int i = 0; i < 4; ++i) v[i] = xr[threadIdx.x + 256 * i];
    float s = v[0] + v[1] + v[2] + v[3];
#pragma unroll
    for (int off = 16; off; off >>= 1) s += __shfl_xor(s, off, 32);
    if ((threadIdx.x & 31) == 0) sred[threadIdx.x >> 5] = s;
    __syncthreads();
    float tot = 0.f;
#pragma unroll
    for (int w = 0; w < 8; ++w) tot += sred[w];
    const float mu = tot * (1.0f / (float)D);
    float q = 0.f;
#pragma unroll
    for (int i = 0; i < 4; ++i) { float d = v[i] - mu; q += d * d; }
#pragma unroll
    for (int off = 16; off; off >>= 1) q += __shfl_xor(q, off, 32);
    __syncthreads();
    if ((threadIdx.x & 31) == 0) sred[threadIdx.x >> 5] = q;
    __syncthreads();
    float qt = 0.f;
#pragma unroll
    for (int w = 0; w < 8; ++w) qt += sred[w];
    const float rstd = rsqrtf(qt * (1.0f / (float)D) + 1e-5f);
#pragma unroll
    for (int i = 0; i < 4; ++i) {
        int c = threadIdx.x + 256 * i;
        out[(size_t)blockIdx.x * D + c] = (__bf16)((v[i] - mu) * rstd * g[c] + be[c]);
    }
}

// ---------------------------------------------------------------------------
// Fragment loaders (this code's contiguous K order, same for A and B)
//   A 16x32 bf16: lane row m0+(l&15); 16 contiguous K from (hi?16:0)
//   B 32x16 bf16: lane col n0+(l&15); 16 contiguous K from (hi?16:0)
// ---------------------------------------------------------------------------
template<int MT>
__device__ __forceinline__ void load_a_frag(const __bf16* const (&Arow)[MT],
                                            int kk, int kaOff, v16bf (&a)[MT]) {
#pragma unroll
    for (int i = 0; i < MT; ++i) {
        v8bf lo = *(const v8bf*)(Arow[i] + kk + kaOff);
        v8bf hs = *(const v8bf*)(Arow[i] + kk + 8 + kaOff);
#pragma unroll
        for (int q = 0; q < 8; ++q) { a[i][q] = lo[q]; a[i][q + 8] = hs[q]; }
    }
}

template<int NT>
__device__ __forceinline__ void load_b_frag(const __bf16* const (&Brow)[NT],
                                            int kk, v16bf (&b)[NT]) {
#pragma unroll
    for (int j = 0; j < NT; ++j) b[j] = *(const v16bf*)(Brow[j] + kk);
}

// ---------------------------------------------------------------------------
// bf16 NT GEMM with WMMA: C[M,N] = A[M,K] * B[N,K]^T (+bias) (+resid) (+act)
// act: 0 = none, 1 = exact GELU, 2 = softplus
// Each wave computes a (MT*16) x (NT*16) strip. Requires:
//   M % (MT*16) == 0, N % (NT*16) == 0, K % 64 == 0.
// K-loop is ping-pong double-buffered so global loads overlap WMMA.
// ---------------------------------------------------------------------------
template<int MT, int NT>
__global__ __launch_bounds__(256)
void gemm_bf16_nt_kernel(const __bf16* __restrict__ A, const __bf16* __restrict__ Bm,
                         const float* __restrict__ bias, const float* __restrict__ resid,
                         float* __restrict__ Cf, __bf16* __restrict__ Cbf,
                         int M, int N, int K, int lda, int ldb, int ldc, int act) {
    const int lane = threadIdx.x & 31;
    const int wave = threadIdx.x >> 5;
    const int nGroups = N / (NT * 16);
    const int mGroups = M / (MT * 16);
    const int unit = blockIdx.x * 8 + wave;
    if (unit >= mGroups * nGroups) return;
    const int m0 = (unit / nGroups) * (MT * 16);
    const int n0 = (unit % nGroups) * (NT * 16);

    const int l15 = lane & 15;
    const int hi  = lane >> 4;      // 0 or 1
    const int kaOff = hi << 4;      // +16 halves for A on upper half-wave
    const int kbOff = hi << 4;      // +16 halves for B on upper half-wave

    const __bf16* Arow[MT];
#pragma unroll
    for (int i = 0; i < MT; ++i)
        Arow[i] = A + (size_t)(m0 + i * 16 + l15) * lda;
    const __bf16* Brow[NT];
#pragma unroll
    for (int j = 0; j < NT; ++j)
        Brow[j] = Bm + (size_t)(n0 + j * 16 + l15) * ldb + kbOff;

    v8f acc[MT][NT];
#pragma unroll
    for (int i = 0; i < MT; ++i)
#pragma unroll
        for (int j = 0; j < NT; ++j) {
            v8f z = {0.f, 0.f, 0.f, 0.f, 0.f, 0.f, 0.f, 0.f};
            acc[i][j] = z;
        }

    v16bf aP[MT], bP[NT], aQ[MT], bQ[NT];
    load_a_frag<MT>(Arow, 0, kaOff, aP);
    load_b_frag<NT>(Brow, 0, bP);

    for (int kk = 0; kk < K; kk += 64) {
        load_a_frag<MT>(Arow, kk + 32, kaOff, aQ);
        load_b_frag<NT>(Brow, kk + 32, bQ);
#pragma unroll
        for (int i = 0; i < MT; ++i)
#pragma unroll
            for (int j = 0; j < NT; ++j)
                acc[i][j] = __builtin_amdgcn_wmma_f32_16x16x32_bf16(
                    false, aP[i], false, bP[j], (short)0, acc[i][j], false, false);
        if (kk + 64 < K) {
            load_a_frag<MT>(Arow, kk + 64, kaOff, aP);
            load_b_frag<NT>(Brow, kk + 64, bP);
        }
#pragma unroll
        for (int i = 0; i < MT; ++i)
#pragma unroll
            for (int j = 0; j < NT; ++j)
                acc[i][j] = __builtin_amdgcn_wmma_f32_16x16x32_bf16(
                    false, aQ[i], false, bQ[j], (short)0, acc[i][j], false, false);
    }

    // Epilogue: C layout -> vgpr r: M = m0+i*16+r+(hi?8:0), N = n0+j*16+(l&15)
#pragma unroll
    for (int i = 0; i < MT; ++i) {
#pragma unroll
        for (int j = 0; j < NT; ++j) {
            const int n = n0 + j * 16 + l15;
            const float bv = bias ? bias[n] : 0.f;
#pragma unroll
            for (int r = 0; r < 8; ++r) {
                const int m = m0 + i * 16 + r + (hi << 3);
                const size_t idx = (size_t)m * ldc + n;
                float v = acc[i][j][r] + bv;
                if (resid) v += resid[idx];
                if (act == 1) v = 0.5f * v * (1.0f + erff(v * 0.70710678118f)); // GELU
                else if (act == 2) v = (v > 20.f) ? v : log1pf(expf(v));        // softplus
                if (Cf)  Cf[idx]  = v;
                if (Cbf) Cbf[idx] = (__bf16)v;
            }
        }
    }
}

// ---------------------------------------------------------------------------
// Causal depthwise conv1d (D_CONV=4) + bias + SiLU on u = xz[:, :, 0:DINNER]
// ---------------------------------------------------------------------------
__global__ __launch_bounds__(256)
void conv_silu_kernel(const float* __restrict__ xz, const float* __restrict__ wconv,
                      const float* __restrict__ bconv,
                      float* __restrict__ u, __bf16* __restrict__ ubf) {
    const int idx = blockIdx.x * 256 + threadIdx.x;
    const int total = BATCH * LSEQ * DINNER;
    if (idx >= total) return;
    const int e = idx % DINNER;
    const int t = (idx / DINNER) % LSEQ;
    const int b = idx / (DINNER * LSEQ);
    const float* __restrict__ col = xz + (size_t)b * LSEQ * TWOE + e;
    float acc = bconv[e];
#pragma unroll
    for (int k = 0; k < DCONV; ++k) {
        const int tt = t - (DCONV - 1) + k;
        if (tt >= 0) acc += wconv[e * DCONV + k] * col[(size_t)tt * TWOE];
    }
    const float s = acc / (1.0f + __expf(-acc));  // SiLU
    u[idx]   = s;
    ubf[idx] = (__bf16)s;
}

// ---------------------------------------------------------------------------
// Chunked parallel selective scan.
// Recurrence per (b,e,n): h_t = exp(dt_t*A_n)*h_{t-1} + dt_t*u_t*B_t[n].
// Chunk transition is affine; the chunk product of decays collapses to
// exp(A_n * sum(dt)) -> only a scalar per (b,e,chunk) plus hpart[16] needed.
// Phase1: per-chunk partial state from h=0 (262144 threads).
// Phase2: serial composition over 64 chunks (4096 threads, 64 steps).
// Phase3: per-chunk replay from true start state + fused gating epilogue.
// ---------------------------------------------------------------------------
__global__ __launch_bounds__(256)
void scan_p1_kernel(const float* __restrict__ dt, const float* __restrict__ xdbl,
                    const float* __restrict__ u, const float* __restrict__ A_log,
                    float* __restrict__ sumdt, float* __restrict__ hpart) {
    const int e = (blockIdx.x & 7) * 256 + threadIdx.x;   // 8 e-blocks
    const int c = (blockIdx.x >> 3) & (NCHUNK - 1);
    const int b = (int)(blockIdx.x >> 9);
    __shared__ float sB[DSTATE];
    float A[DSTATE], h[DSTATE];
#pragma unroll
    for (int n = 0; n < DSTATE; ++n) {
        A[n] = -__expf(A_log[e * DSTATE + n]);
        h[n] = 0.f;
    }
    float sdt = 0.f;
    for (int tt = 0; tt < CLEN; ++tt) {
        const size_t row = (size_t)b * LSEQ + c * CLEN + tt;
        if (threadIdx.x < DSTATE)
            sB[threadIdx.x] = xdbl[row * NXDBL + DTRANK + threadIdx.x];
        __syncthreads();
        const float dtv = dt[row * DINNER + e];
        const float du  = dtv * u[row * DINNER + e];
        sdt += dtv;
#pragma unroll
        for (int n = 0; n < DSTATE; ++n)
            h[n] = __expf(dtv * A[n]) * h[n] + du * sB[n];
        __syncthreads();
    }
    const size_t base = ((size_t)(b * NCHUNK + c)) * DINNER + e;
    sumdt[base] = sdt;
#pragma unroll
    for (int n = 0; n < DSTATE; ++n) hpart[base * DSTATE + n] = h[n];
}

__global__ __launch_bounds__(256)
void scan_p2_kernel(const float* __restrict__ sumdt, const float* __restrict__ hpart,
                    const float* __restrict__ A_log, float* __restrict__ h0st) {
    const int gid = blockIdx.x * 256 + threadIdx.x;
    const int e = gid & (DINNER - 1);
    const int b = gid >> 11;
    float A[DSTATE], h[DSTATE];
#pragma unroll
    for (int n = 0; n < DSTATE; ++n) {
        A[n] = -__expf(A_log[e * DSTATE + n]);
        h[n] = 0.f;
    }
    for (int c = 0; c < NCHUNK; ++c) {
        const size_t base = ((size_t)(b * NCHUNK + c)) * DINNER + e;
#pragma unroll
        for (int n = 0; n < DSTATE; ++n) h0st[base * DSTATE + n] = h[n];
        const float sd = sumdt[base];
#pragma unroll
        for (int n = 0; n < DSTATE; ++n)
            h[n] = __expf(A[n] * sd) * h[n] + hpart[base * DSTATE + n];
    }
}

__global__ __launch_bounds__(256)
void scan_p3_kernel(const float* __restrict__ dt, const float* __restrict__ xdbl,
                    const float* __restrict__ u, const float* __restrict__ xz,
                    const float* __restrict__ A_log, const float* __restrict__ Dp,
                    const float* __restrict__ h0st, __bf16* __restrict__ ybf) {
    const int e = (blockIdx.x & 7) * 256 + threadIdx.x;
    const int c = (blockIdx.x >> 3) & (NCHUNK - 1);
    const int b = (int)(blockIdx.x >> 9);
    __shared__ float sBC[2 * DSTATE];
    float A[DSTATE], h[DSTATE];
    const size_t base = ((size_t)(b * NCHUNK + c)) * DINNER + e;
#pragma unroll
    for (int n = 0; n < DSTATE; ++n) {
        A[n] = -__expf(A_log[e * DSTATE + n]);
        h[n] = h0st[base * DSTATE + n];
    }
    const float Dpe = Dp[e];
    for (int tt = 0; tt < CLEN; ++tt) {
        const size_t row = (size_t)b * LSEQ + c * CLEN + tt;
        if (threadIdx.x < 2 * DSTATE)
            sBC[threadIdx.x] = xdbl[row * NXDBL + DTRANK + threadIdx.x];
        __syncthreads();
        const float dtv = dt[row * DINNER + e];
        const float uv  = u[row * DINNER + e];
        const float du  = dtv * uv;
        float y = 0.f;
#pragma unroll
        for (int n = 0; n < DSTATE; ++n) {
            h[n] = __expf(dtv * A[n]) * h[n] + du * sBC[n];
            y += h[n] * sBC[DSTATE + n];
        }
        const float z = xz[row * TWOE + DINNER + e];
        ybf[row * DINNER + e] = (__bf16)((y + uv * Dpe) * (z / (1.0f + __expf(-z))));
        __syncthreads();
    }
}

// ---------------------------------------------------------------------------
// Host orchestration
// ---------------------------------------------------------------------------
extern "C" void kernel_launch(void* const* d_in, const int* in_sizes, int n_in,
                              void* d_out, int out_size, void* d_ws, size_t ws_size,
                              hipStream_t stream) {
    const float* x     = (const float*)d_in[0];
    const float* g1    = (const float*)d_in[1];
    const float* be1   = (const float*)d_in[2];
    const float* Win   = (const float*)d_in[3];
    const float* Wconv = (const float*)d_in[4];
    const float* bconv = (const float*)d_in[5];
    const float* Wx    = (const float*)d_in[6];
    const float* Wdt   = (const float*)d_in[7];
    const float* bdt   = (const float*)d_in[8];
    const float* A_log = (const float*)d_in[9];
    const float* Dp    = (const float*)d_in[10];
    const float* Wout  = (const float*)d_in[11];
    const float* g2    = (const float*)d_in[12];
    const float* be2   = (const float*)d_in[13];
    const float* W1    = (const float*)d_in[14];
    const float* bf1   = (const float*)d_in[15];
    const float* W2    = (const float*)d_in[16];
    const float* bf2   = (const float*)d_in[17];
    float* out = (float*)d_out;

    char* ws = (char*)d_ws;
    size_t off = 0;
    auto alloc = [&](size_t bytes) -> void* {
        void* p = ws + off;
        off += (bytes + 255) & ~(size_t)255;
        return p;
    };

    __bf16* Winbf  = (__bf16*)alloc((size_t)TWOE * DMODEL * 2);
    __bf16* Wxbf   = (__bf16*)alloc((size_t)NXDBL * DINNER * 2);
    __bf16* Wdtbf  = (__bf16*)alloc((size_t)DINNER * DTRANK * 2);
    __bf16* Woutbf = (__bf16*)alloc((size_t)DMODEL * DINNER * 2);
    __bf16* W1bf   = (__bf16*)alloc((size_t)FFH * DMODEL * 2);
    __bf16* W2bf   = (__bf16*)alloc((size_t)DMODEL * FFH * 2);
    __bf16* xnbf   = (__bf16*)alloc((size_t)ROWS * DMODEL * 2);   // LN1 out, reused for LN2
    float*  xzf    = (float*)alloc((size_t)ROWS * TWOE * 4);
    float*  uf     = (float*)alloc((size_t)ROWS * DINNER * 4);
    __bf16* ubf    = (__bf16*)alloc((size_t)ROWS * DINNER * 2);
    float*  xdblf  = (float*)alloc((size_t)ROWS * NXDBL * 4);
    __bf16* xdblbf = (__bf16*)alloc((size_t)ROWS * NXDBL * 2);
    float*  dtf    = (float*)alloc((size_t)ROWS * DINNER * 4);
    __bf16* ybf    = (__bf16*)alloc((size_t)ROWS * DINNER * 2);
    float*  x1f    = (float*)alloc((size_t)ROWS * DMODEL * 4);
    __bf16* hbf    = (__bf16*)alloc((size_t)ROWS * FFH * 2);
    float*  sumdt  = (float*)alloc((size_t)BATCH * NCHUNK * DINNER * 4);
    float*  hpart  = (float*)alloc((size_t)BATCH * NCHUNK * DINNER * DSTATE * 4);
    float*  h0st   = (float*)alloc((size_t)BATCH * NCHUNK * DINNER * DSTATE * 4);
    (void)ws_size; (void)n_in; (void)in_sizes; (void)out_size;

    auto cvt = [&](const float* src, __bf16* dst, int n) {
        cvt_bf16_kernel<<<(n + 255) / 256, 256, 0, stream>>>(src, dst, n);
    };
    // wave tile = 32x64
    auto gemm4 = [&](const __bf16* A, const __bf16* Bm, const float* bias,
                     const float* resid, float* Cf, __bf16* Cbf,
                     int M, int N, int K, int lda, int ldb, int ldc, int act) {
        const int units = (M / 32) * (N / 64);
        gemm_bf16_nt_kernel<2, 4><<<(units + 7) / 8, 256, 0, stream>>>(
            A, Bm, bias, resid, Cf, Cbf, M, N, K, lda, ldb, ldc, act);
    };
    // wave tile = 32x32 (for N=96)
    auto gemm2 = [&](const __bf16* A, const __bf16* Bm, const float* bias,
                     const float* resid, float* Cf, __bf16* Cbf,
                     int M, int N, int K, int lda, int ldb, int ldc, int act) {
        const int units = (M / 32) * (N / 32);
        gemm_bf16_nt_kernel<2, 2><<<(units + 7) / 8, 256, 0, stream>>>(
            A, Bm, bias, resid, Cf, Cbf, M, N, K, lda, ldb, ldc, act);
    };

    // 1) weights -> bf16 (deterministic, every call)
    cvt(Win,  Winbf,  TWOE * DMODEL);
    cvt(Wx,   Wxbf,   NXDBL * DINNER);
    cvt(Wdt,  Wdtbf,  DINNER * DTRANK);
    cvt(Wout, Woutbf, DMODEL * DINNER);
    cvt(W1,   W1bf,   FFH * DMODEL);
    cvt(W2,   W2bf,   DMODEL * FFH);

    // 2) LN1 -> bf16
    ln_bf16_kernel<<<ROWS, 256, 0, stream>>>(x, g1, be1, xnbf);

    // 3) xz = xn @ Win^T   [4096 x 4096], K=1024
    gemm4(xnbf, Winbf, nullptr, nullptr, xzf, nullptr,
          ROWS, TWOE, DMODEL, DMODEL, DMODEL, TWOE, 0);

    // 4) causal depthwise conv + SiLU on u half
    conv_silu_kernel<<<(ROWS * DINNER + 255) / 256, 256, 0, stream>>>(
        xzf, Wconv, bconv, uf, ubf);

    // 5) x_dbl = u @ Wx^T   [4096 x 96], K=2048; f32 (scan B/C) + bf16 (dt GEMM A)
    gemm2(ubf, Wxbf, nullptr, nullptr, xdblf, xdblbf,
          ROWS, NXDBL, DINNER, DINNER, DINNER, NXDBL, 0);

    // 6) dt = softplus(dt_in @ Wdt^T + bdt)   [4096 x 2048], K=64
    gemm4(xdblbf, Wdtbf, bdt, nullptr, dtf, nullptr,
          ROWS, DINNER, DTRANK, NXDBL, DTRANK, DINNER, 2);

    // 7) chunked parallel selective scan + fused gating epilogue -> ybf
    scan_p1_kernel<<<BATCH * NCHUNK * (DINNER / 256), 256, 0, stream>>>(
        dtf, xdblf, uf, A_log, sumdt, hpart);
    scan_p2_kernel<<<(BATCH * DINNER) / 256, 256, 0, stream>>>(
        sumdt, hpart, A_log, h0st);
    scan_p3_kernel<<<BATCH * NCHUNK * (DINNER / 256), 256, 0, stream>>>(
        dtf, xdblf, uf, xzf, A_log, Dp, h0st, ybf);

    // 8) x1 = x + y @ Wout^T   [4096 x 1024], K=2048
    gemm4(ybf, Woutbf, nullptr, x, x1f, nullptr,
          ROWS, DMODEL, DINNER, DINNER, DINNER, DMODEL, 0);

    // 9) LN2 -> bf16 (reuse xnbf)
    ln_bf16_kernel<<<ROWS, 256, 0, stream>>>(x1f, g2, be2, xnbf);

    // 10) h = gelu(xn2 @ W1^T + bf1) -> bf16   [4096 x 4096], K=1024
    gemm4(xnbf, W1bf, bf1, nullptr, nullptr, hbf,
          ROWS, FFH, DMODEL, DMODEL, DMODEL, FFH, 1);

    // 11) out = x1 + h @ W2^T + bf2   [4096 x 1024], K=4096
    gemm4(hbf, W2bf, bf2, x1f, out, nullptr,
          ROWS, DMODEL, FFH, FFH, FFH, DMODEL, 0);
}
